// CustomWindowMHA_39591008534780
// MI455X (gfx1250) — matrix operand order repaired and checked by reading, not compiled, not run on hardware
//
#include <hip/hip_runtime.h>

// ---------------------------------------------------------------------------
// Sparse window+dilated MHA for MI455X (gfx1250, wave32, WMMA + async/TDM).
// B=2, S=2048, D=1024, H=16, dh=64, WINDOW=128, DILATION=4.
//  1) f32 -> bf16 convert
//  2) QKV GEMM: WMMA bf16, LDS double-buffered via global_load_async_to_lds
//  3) flash attention, per-element sparse mask, K/V tiles staged with TDM,
//     softmax row-reductions via DPP ROW_XMASK, P transpose via LDS
//  4) output GEMM (same async-staged WMMA GEMM), f32 store
// ---------------------------------------------------------------------------

typedef __attribute__((ext_vector_type(16))) __bf16    v16bf;
typedef __attribute__((ext_vector_type(8)))  float     v8f;
typedef __attribute__((ext_vector_type(4)))  unsigned  v4u;
typedef __attribute__((ext_vector_type(8)))  int       v8i;
typedef __attribute__((ext_vector_type(4)))  int       v4i;

#define SEQ   2048
#define NH    16
#define DHEAD 64
#define DM    1024

static __device__ __forceinline__ unsigned short f2bfbits(float f) {
    union { float f; unsigned u; } v; v.f = f;
    unsigned u = v.u;
    u += 0x7FFFu + ((u >> 16) & 1u);   // round-to-nearest-even
    return (unsigned short)(u >> 16);
}

// Build a 16-half fragment from two 16-byte chunks (global or LDS pointers).
static __device__ __forceinline__ v16bf load_frag16(const unsigned short* p0,
                                                    const unsigned short* p1) {
    v16bf r;
    float4* rp = reinterpret_cast<float4*>(&r);
    rp[0] = *reinterpret_cast<const float4*>(p0);
    rp[1] = *reinterpret_cast<const float4*>(p1);
    return r;
}

// Async 16B copy global -> LDS (ASYNCcnt-tracked, no VGPR round trip).
static __device__ __forceinline__ void async_load_b128(unsigned lds_off, const void* g) {
    asm volatile("global_load_async_to_lds_b128 %0, %1, off"
                 :: "v"(lds_off), "v"((unsigned long long)(size_t)g)
                 : "memory");
}

// TDM: 2-D bf16 tile (tile_d1 rows x tile_d0 elems, row stride = stride0 elems)
// from global into LDS at lds_off. D# per CDNA5 ISA 8.3/8.4 (count=1, type=2,
// data_size=1 -> 2 bytes). Groups 2/3 zero (<=2-D tensor).
static __device__ __forceinline__ void tdm_load_2d(unsigned lds_off, const void* gaddr,
                                                   unsigned tensor_d0, unsigned tensor_d1,
                                                   unsigned tile_d0, unsigned tile_d1,
                                                   unsigned stride0) {
    unsigned long long ga = (unsigned long long)(size_t)gaddr;
    v4u g0;
    g0.x = 1u;                                   // count=1 (valid), user mode
    g0.y = lds_off;                              // lds_addr (bytes)
    g0.z = (unsigned)(ga & 0xFFFFFFFFu);         // global_addr[31:0]
    g0.w = (unsigned)((ga >> 32) & 0x01FFFFFFu)  // global_addr[56:32]
         | (2u << 30);                           // type = 2 ("image")
    v8i g1;
    g1[0] = (int)(1u << 16);                                                  // data_size=1 (2B)
    g1[1] = (int)((tensor_d0 & 0xFFFFu) << 16);                               // tensor_dim0[15:0]
    g1[2] = (int)(((tensor_d0 >> 16) & 0xFFFFu) | ((tensor_d1 & 0xFFFFu) << 16));
    g1[3] = (int)(((tensor_d1 >> 16) & 0xFFFFu) | ((tile_d0 & 0xFFFFu) << 16));
    g1[4] = (int)(tile_d1 & 0xFFFFu);                                         // tile_dim1, tile_dim2=0
    g1[5] = (int)stride0;                                                     // dim0_stride[31:0]
    g1[6] = 0;
    g1[7] = 0;
    v4i z4 = {0, 0, 0, 0};
    v8i z8 = {0, 0, 0, 0, 0, 0, 0, 0};
    __builtin_amdgcn_tensor_load_to_lds(g0, g1, z4, z4, z8, 0);
}

// 16-lane (half-wave) butterfly reductions via DPP ROW_XMASK (stays in VALU).
#define DPP_XOR(dst, src, mask)                                              \
    dst = __builtin_bit_cast(float, __builtin_amdgcn_update_dpp(             \
              __builtin_bit_cast(int, src), __builtin_bit_cast(int, src),    \
              0x160 | (mask), 0xf, 0xf, false))

static __device__ __forceinline__ float rowmax16(float x) {
    float t;
    DPP_XOR(t, x, 1); x = fmaxf(x, t);
    DPP_XOR(t, x, 2); x = fmaxf(x, t);
    DPP_XOR(t, x, 4); x = fmaxf(x, t);
    DPP_XOR(t, x, 8); x = fmaxf(x, t);
    return x;
}
static __device__ __forceinline__ float rowsum16(float x) {
    float t;
    DPP_XOR(t, x, 1); x += t;
    DPP_XOR(t, x, 2); x += t;
    DPP_XOR(t, x, 4); x += t;
    DPP_XOR(t, x, 8); x += t;
    return x;
}

__global__ __launch_bounds__(256)
void convert_f32_bf16(const float* __restrict__ in, unsigned short* __restrict__ out, int n) {
    int i = blockIdx.x * blockDim.x + threadIdx.x;
    if (i < n) out[i] = f2bfbits(in[i]);
}

// ---------------------------------------------------------------------------
// GEMM: C[M,N] = A[M,K] (bf16 rm) * Bw[N,K]^T (bf16 rm).
// Block 256 thr = 8 waves; block tile 64x256; wave tile 32x64 (2x4 wmma).
// K-steps of 32 staged in LDS (A 64x32, B 256x32), double-buffered with
// global_load_async_to_lds_b128 + s_wait_asynccnt, pipelined one step ahead.
// MODE 0: f32 store to outF.  MODE 1: scatter Q,K bf16 + V transposed bf16.
// ---------------------------------------------------------------------------
template <int MODE>
__global__ __launch_bounds__(256)
void gemm_bf16_wmma(const unsigned short* __restrict__ A,
                    const unsigned short* __restrict__ Bw,
                    int M, int N, int K,
                    float* __restrict__ outF,
                    unsigned short* __restrict__ qb,
                    unsigned short* __restrict__ kb,
                    unsigned short* __restrict__ vtb)
{
    __shared__ unsigned short shA[2][64 * 32];    // 2 x 4KB
    __shared__ unsigned short shB[2][256 * 32];   // 2 x 16KB

    const int tid  = threadIdx.x;
    const int lane = tid & 31;
    const int w    = tid >> 5;
    const int wm = w & 1, wn = w >> 1;
    const int m_blk = blockIdx.y * 64;
    const int n_blk = blockIdx.x * 256;
    const int lr = lane & 15;
    const int hi = lane >> 4;

    // per-thread async-copy assignment: A tile 256x16B chunks (1/thread),
    // B tile 1024x16B chunks (4/thread) -> 5 async ops per thread per step.
    const int a_row = tid >> 2;
    const int a_sub = tid & 3;

    auto issue = [&](int c, int kk) {
        {
            unsigned loff = (unsigned)(size_t)&shA[c][a_row * 32 + a_sub * 8];
            async_load_b128(loff, A + (size_t)(m_blk + a_row) * K + kk + a_sub * 8);
        }
        #pragma unroll
        for (int i = 0; i < 4; i++) {
            int id  = i * 256 + tid;
            int row = id >> 2, sub = id & 3;
            unsigned loff = (unsigned)(size_t)&shB[c][row * 32 + sub * 8];
            async_load_b128(loff, Bw + (size_t)(n_blk + row) * K + kk + sub * 8);
        }
    };

    v8f acc[2][4];
    #pragma unroll
    for (int i = 0; i < 2; i++)
        #pragma unroll
        for (int j = 0; j < 4; j++) acc[i][j] = {};

    const int ksteps = K >> 5;
    issue(0, 0);
    int c = 0;
    for (int ks = 0; ks < ksteps; ks++) {
        if (ks + 1 < ksteps) {
            issue(c ^ 1, (ks + 1) << 5);                       // prefetch next step
            asm volatile("s_wait_asynccnt 0x5" ::: "memory");  // this step's 5 done
        } else {
            asm volatile("s_wait_asynccnt 0x0" ::: "memory");
        }
        __syncthreads();

        v16bf af[2];
        #pragma unroll
        for (int mi = 0; mi < 2; mi++) {
            const unsigned short* p = &shA[c][(wm * 32 + mi * 16 + lr) * 32 + hi * 8];
            af[mi] = load_frag16(p, p + 16);
        }
        v16bf bfr[4];
        #pragma unroll
        for (int ni = 0; ni < 4; ni++) {
            const unsigned short* p = &shB[c][(wn * 64 + ni * 16 + lr) * 32 + hi * 8];
            bfr[ni] = load_frag16(p, p + 16);
        }
        #pragma unroll
        for (int mi = 0; mi < 2; mi++)
            #pragma unroll
            for (int ni = 0; ni < 4; ni++)
                acc[mi][ni] = __builtin_amdgcn_wmma_f32_16x16x32_bf16(
                    false, af[mi], false, bfr[ni], (short)0, acc[mi][ni], false, false);
        __syncthreads();
        c ^= 1;
    }

    // Epilogue. C layout: vgpr r, lanes 0-15 -> M=r, lanes 16-31 -> M=8+r.
    #pragma unroll
    for (int mi = 0; mi < 2; mi++) {
        #pragma unroll
        for (int ni = 0; ni < 4; ni++) {
            #pragma unroll
            for (int r = 0; r < 8; r++) {
                int m = m_blk + wm * 32 + mi * 16 + r + hi * 8;
                int n = n_blk + wn * 64 + ni * 16 + lr;
                float val = acc[mi][ni][r];
                if (MODE == 0) {
                    outF[(size_t)m * N + n] = val;
                } else {
                    int which = n >> 10;          // 0:Q 1:K 2:V
                    int cch = n & 1023;
                    int h = cch >> 6, d = cch & 63;
                    unsigned short bits = f2bfbits(val);
                    if (which == 0) {
                        qb[((size_t)m * NH + h) * DHEAD + d] = bits;
                    } else if (which == 1) {
                        kb[((size_t)m * NH + h) * DHEAD + d] = bits;
                    } else {
                        int bidx = m >> 11, s = m & 2047;
                        vtb[(((size_t)bidx * NH + h) * DHEAD + d) * SEQ + s] = bits;
                    }
                }
            }
        }
    }
}

// ---------------------------------------------------------------------------
// Flash attention, one wave per (b, h, 16-query tile), 8 waves/block.
// K tile (32 keys x 64 dh) and V^T tile (64 dh x 32 keys) staged per-wave
// with TDM; softmax reductions via DPP; P transpose via per-wave LDS.
// ---------------------------------------------------------------------------
__global__ __launch_bounds__(256)
void attn_sparse_wmma(const unsigned short* __restrict__ qb,
                      const unsigned short* __restrict__ kbuf,
                      const unsigned short* __restrict__ vtb,
                      unsigned short* __restrict__ aout)
{
    __shared__ unsigned short shK[8][32 * 64];   // per-wave 4KB: rows=keys, cols=dh
    __shared__ unsigned short shV[8][64 * 32];   // per-wave 4KB: rows=dh,  cols=keys
    __shared__ unsigned short shP[8][16 * 32];   // per-wave 1KB: P transpose bounce

    const int lane = threadIdx.x & 31;
    const int w    = threadIdx.x >> 5;
    const int lr = lane & 15;
    const int hi = lane >> 4;

    int tile = blockIdx.x * 8 + w;   // 0..4095
    int qt   = tile & 127;
    int head = tile >> 7;
    int b = head >> 4;
    int h = head & 15;
    int i0 = qt << 4;

    unsigned short* kw = &shK[w][0];
    unsigned short* vw = &shV[w][0];
    unsigned short* pw = &shP[w][0];
    const unsigned kldsoff = (unsigned)(size_t)kw;
    const unsigned vldsoff = (unsigned)(size_t)vw;

    // Q A-fragments (16x32 each), dh-chunks [0,32) and [32,64)
    v16bf qf[2];
    {
        const unsigned short* qrow = qb + ((size_t)(b * SEQ + i0 + lr) * NH + h) * DHEAD;
        #pragma unroll
        for (int c = 0; c < 2; c++) {
            const unsigned short* p = qrow + c * 32 + hi * 8;
            qf[c] = load_frag16(p, p + 16);
        }
    }

    v8f acc[4];
    #pragma unroll
    for (int t = 0; t < 4; t++) acc[t] = {};
    float mrow[8], lsum[8];
    #pragma unroll
    for (int r = 0; r < 8; r++) { mrow[r] = -1e30f; lsum[r] = 0.f; }

    const unsigned short* vbase = vtb + ((size_t)(b * NH + h) * DHEAD) * SEQ;

    for (int j0 = 0; j0 < i0 + 16; j0 += 32) {
        // ---- stage K (32x64) and V^T (64x32) tiles via Tensor Data Mover ----
        asm volatile("s_wait_dscnt 0x0" ::: "memory");   // prior LDS reads retired
        tdm_load_2d(kldsoff, kbuf + ((size_t)(b * SEQ + j0) * NH + h) * DHEAD,
                    DHEAD, SEQ, DHEAD, 32, NH * DHEAD);
        tdm_load_2d(vldsoff, vbase + j0,
                    SEQ, DHEAD, 32, DHEAD, SEQ);
        __builtin_amdgcn_s_wait_tensorcnt(0);
        asm volatile("" ::: "memory");

        // ---- scores: two 16x16 tiles (keys j0..j0+15, j0+16..j0+31) ----
        v8f s0 = {}, s1 = {};
        #pragma unroll
        for (int c = 0; c < 2; c++) {
            const unsigned short* k0 = kw + (size_t)(lr)      * DHEAD + c * 32 + hi * 8;
            const unsigned short* k1 = kw + (size_t)(16 + lr) * DHEAD + c * 32 + hi * 8;
            v16bf kf0 = load_frag16(k0, k0 + 16);
            v16bf kf1 = load_frag16(k1, k1 + 16);
            s0 = __builtin_amdgcn_wmma_f32_16x16x32_bf16(false, qf[c], false, kf0, (short)0, s0, false, false);
            s1 = __builtin_amdgcn_wmma_f32_16x16x32_bf16(false, qf[c], false, kf1, (short)0, s1, false, false);
        }

        // ---- scale + sparse mask (window OR dilated-causal) ----
        float p0[8], p1[8], alpha[8];
        #pragma unroll
        for (int r = 0; r < 8; r++) {
            int irow = i0 + r + hi * 8;
            int jc0 = j0 + lr, jc1 = jc0 + 16;
            float v0 = s0[r] * 0.125f;
            float v1 = s1[r] * 0.125f;
            bool a0 = (jc0 <= irow) && ((jc0 >= irow - 127) ||
                      ((jc0 <= irow - 132) && (((irow - jc0) & 3) == 0)));
            bool a1 = (jc1 <= irow) && ((jc1 >= irow - 127) ||
                      ((jc1 <= irow - 132) && (((irow - jc1) & 3) == 0)));
            p0[r] = a0 ? v0 : -1e30f;
            p1[r] = a1 ? v1 : -1e30f;
        }
        #pragma unroll
        for (int r = 0; r < 8; r++) {
            float rmx = rowmax16(fmaxf(p0[r], p1[r]));
            float mnew = fmaxf(mrow[r], rmx);
            alpha[r] = __expf(mrow[r] - mnew);
            p0[r]    = __expf(p0[r] - mnew);
            p1[r]    = __expf(p1[r] - mnew);
            mrow[r]  = mnew;
            float rs = rowsum16(p0[r] + p1[r]);
            lsum[r] = lsum[r] * alpha[r] + rs;
        }
        #pragma unroll
        for (int t = 0; t < 4; t++)
            #pragma unroll
            for (int r = 0; r < 8; r++)
                acc[t][r] *= alpha[r];

        // ---- P (C-layout) -> LDS row-major 16x32 bf16 ----
        #pragma unroll
        for (int r = 0; r < 8; r++) {
            int rl = r + hi * 8;
            pw[rl * 32 + lr]      = f2bfbits(p0[r]);
            pw[rl * 32 + lr + 16] = f2bfbits(p1[r]);
        }
        asm volatile("s_wait_dscnt 0x0" ::: "memory");

        // ---- reload P as A-fragment (16x32) ----
        const unsigned short* pp = pw + lr * 32 + hi * 8;
        v16bf pf = load_frag16(pp, pp + 16);

        // ---- PV: 4 dh tiles from staged V^T ----
        #pragma unroll
        for (int t = 0; t < 4; t++) {
            const unsigned short* vp = vw + (size_t)(t * 16 + lr) * 32 + hi * 8;
            v16bf vf = load_frag16(vp, vp + 16);
            acc[t] = __builtin_amdgcn_wmma_f32_16x16x32_bf16(
                false, pf, false, vf, (short)0, acc[t], false, false);
        }
    }

    // ---- normalize and store bf16 [B,S,H*dh] ----
    #pragma unroll
    for (int r = 0; r < 8; r++) {
        float inv = 1.f / lsum[r];
        int irow = i0 + r + hi * 8;
        size_t rowoff = ((size_t)(b * SEQ + irow)) * DM + h * DHEAD;
        #pragma unroll
        for (int t = 0; t < 4; t++)
            aout[rowoff + t * 16 + lr] = f2bfbits(acc[t][r] * inv);
    }
}

// ---------------------------------------------------------------------------
extern "C" void kernel_launch(void* const* d_in, const int* in_sizes, int n_in,
                              void* d_out, int out_size, void* d_ws, size_t ws_size,
                              hipStream_t stream) {
    const float* x   = (const float*)d_in[0];   // [2,2048,1024]
    const float* qkv = (const float*)d_in[1];   // [3072,1024]
    const float* wo  = (const float*)d_in[2];   // [1024,1024]
    float* outF = (float*)d_out;                // [2,2048,1024]

    const int M  = 2 * SEQ;          // 4096 tokens
    const int N1 = 3 * DM;           // 3072
    const int K  = DM;               // 1024

    unsigned short* xb    = (unsigned short*)d_ws;                 // 4096*1024
    unsigned short* qkvb  = xb    + (size_t)M * DM;                // 3072*1024
    unsigned short* wob   = qkvb  + (size_t)N1 * DM;               // 1024*1024
    unsigned short* qbuf  = wob   + (size_t)DM * DM;               // 4096*1024
    unsigned short* kbuf  = qbuf  + (size_t)M * DM;                // 4096*1024
    unsigned short* vtbuf = kbuf  + (size_t)M * DM;                // 2*16*64*2048
    unsigned short* aout  = vtbuf + (size_t)M * DM;                // 4096*1024

    {
        int n;
        n = M * DM;
        convert_f32_bf16<<<(n + 255) / 256, 256, 0, stream>>>(x, xb, n);
        n = N1 * DM;
        convert_f32_bf16<<<(n + 255) / 256, 256, 0, stream>>>(qkv, qkvb, n);
        n = DM * DM;
        convert_f32_bf16<<<(n + 255) / 256, 256, 0, stream>>>(wo, wob, n);
    }

    {   // QKV projection + Q/K/Vt scatter
        dim3 grid(N1 / 256, M / 64);
        gemm_bf16_wmma<1><<<grid, 256, 0, stream>>>(xb, qkvb, M, N1, K,
                                                    nullptr, qbuf, kbuf, vtbuf);
    }

    {   // sparse flash attention: 4096 wave-tiles, 8 per block
        attn_sparse_wmma<<<512, 256, 0, stream>>>(qbuf, kbuf, vtbuf, aout);
    }

    {   // output projection
        dim3 grid(DM / 256, M / 64);
        gemm_bf16_wmma<0><<<grid, 256, 0, stream>>>(aout, wob, M, DM, K,
                                                    outF, nullptr, nullptr, nullptr);
    }
}
